// ConformerEncoder_77206332113678
// MI455X (gfx1250) — matrix operand order repaired and checked by reading, not compiled, not run on hardware
//
#include <hip/hip_runtime.h>

// ---------------------------------------------------------------------------
// Conformer encoder forward for MI455X (gfx1250), wave32 + WMMA bf16.
// B=8, T=512, D=512, H=8, DK=64, L=4, K=31, DFF=2048.
// ---------------------------------------------------------------------------

#define DEV __device__ __forceinline__

typedef __attribute__((ext_vector_type(16))) __bf16 v16bf;
typedef __attribute__((ext_vector_type(8)))  float  v8f;

constexpr int Bc   = 8;
constexpr int Tc   = 512;
constexpr int Dc   = 512;
constexpr int Hc   = 8;
constexpr int DKc  = 64;
constexpr int DFFc = 2048;
constexpr int KWc  = 31;
constexpr int ROWS = Bc * Tc;          // 4096
constexpr int PADW = (KWc - 1) / 2;    // 15

DEV v8f wmma_bf16(v16bf a, v16bf b, v8f c) {
  // (neg_a, A, neg_b, B, c_mod, C, reuse_a, reuse_b)
  return __builtin_amdgcn_wmma_f32_16x16x32_bf16(false, a, false, b, (short)0, c,
                                                 false, false);
}
DEV float sigmoidf_(float x) { return 1.f / (1.f + __expf(-x)); }
DEV float swishf_(float x)   { return x * sigmoidf_(x); }

DEV float4 ld4(const float* p) { return *(const float4*)p; }
DEV void put4(v16bf& f, int base, float4 x) {
  f[base + 0] = (__bf16)x.x; f[base + 1] = (__bf16)x.y;
  f[base + 2] = (__bf16)x.z; f[base + 3] = (__bf16)x.w;
}
// Load 16 contiguous floats at p into bf16 fragment slots [0..15].
DEV void put16(v16bf& f, const float* p) {
  put4(f, 0, ld4(p)); put4(f, 4, ld4(p + 4));
  put4(f, 8, ld4(p + 8)); put4(f, 12, ld4(p + 12));
}
// A-fragment: K runs are 8-float spans at +hi*16 and +hi*16+8.
DEV void putA(v16bf& f, const float* rowk0, int hi) {
  const float* p0 = rowk0 + hi * 16;
  put4(f, 0, ld4(p0)); put4(f, 4, ld4(p0 + 4));
  const float* p1 = rowk0 + hi * 16 + 8;
  put4(f, 8, ld4(p1)); put4(f, 12, ld4(p1 + 4));
}

// Async global->LDS 16-byte copy (CDNA5, tracked by ASYNCcnt).
// ldsByteOff is the byte offset within the workgroup's LDS allocation.
DEV void async_g2l_b128(unsigned ldsByteOff, const float* gsrc) {
  asm volatile("global_load_async_to_lds_b128 %0, %1, off"
               :: "v"(ldsByteOff), "v"(gsrc) : "memory");
}
DEV void wait_asynccnt0() {
  asm volatile("s_wait_asynccnt 0" ::: "memory");
}

// ---------------------------------------------------------------------------
// WMMA GEMM: out[M,N] = epi(A[M,Kd] * B[Kd,N] + bias)
//  - 128 threads (4 waves), 64x64 tile: wave w owns rows [m0+16w, m0+16w+16)
//  - B tile (32 x 64) staged in LDS; non-transposed path uses CDNA5
//    global_load_async_to_lds_b128 (sB is the only LDS object -> offset 0)
//  - ACT==1: swish; RESID: out = resid + resScale*val
// ---------------------------------------------------------------------------
template<bool TRANSB, int ACT, bool RESID>
__global__ __launch_bounds__(128) void k_gemm(
    const float* __restrict__ A, int lda,
    const float* __restrict__ Bm, int ldb,
    const float* __restrict__ bias,
    const float* __restrict__ resid, float resScale,
    float* __restrict__ out, int M, int N, int Kd)
{
  __shared__ float sB[32][65];           // 32 k x 64 n (+1 pad); LDS offset 0

  const int tid  = threadIdx.x;
  const int wave = tid >> 5;
  const int lane = tid & 31;
  const int hi = lane >> 4;
  const int lm = lane & 15;
  const int m0 = blockIdx.x * 64 + wave * 16;
  const int n0 = blockIdx.y * 64;

  int row = m0 + lm;
  if (row >= M) row = M - 1;             // clamp for ragged M (pos proj: 1023)
  const float* __restrict__ arow = A + (size_t)row * lda;

  v8f acc[4] = {v8f{}, v8f{}, v8f{}, v8f{}};

  for (int k0 = 0; k0 < Kd; k0 += 32) {
    __syncthreads();                     // protect previous iteration's reads
    // ---- stage B tile into LDS ----
    if (!TRANSB) {
      // memory rows are k (contiguous along n): direct async global->LDS DMA
#pragma unroll
      for (int i = 0; i < 4; ++i) {
        const int linear = tid + i * 128;       // 0..511
        const int r  = linear >> 4;             // k row 0..31
        const int c4 = (linear & 15) << 2;      // n col 0..60
        const float* src = &Bm[(size_t)(k0 + r) * ldb + n0 + c4];
        if (k0 + 32 < Kd) __builtin_prefetch(src + (size_t)32 * ldb, 0, 1);
        async_g2l_b128((unsigned)((r * 65 + c4) * sizeof(float)), src);
      }
      wait_asynccnt0();                  // our lanes' async copies done
    } else {
      // memory rows are n (contiguous along k): load + transpose into LDS
#pragma unroll
      for (int i = 0; i < 4; ++i) {
        const int linear = tid + i * 128;
        const int n  = linear >> 3;             // 0..63
        const int k4 = (linear & 7) << 2;       // 0..28
        const float* src = &Bm[(size_t)(n0 + n) * ldb + k0 + k4];
        if (k0 + 32 < Kd) __builtin_prefetch(src + 32, 0, 1);
        const float4 v = ld4(src);
        sB[k4 + 0][n] = v.x; sB[k4 + 1][n] = v.y;
        sB[k4 + 2][n] = v.z; sB[k4 + 3][n] = v.w;
      }
    }
    __syncthreads();

    // ---- A fragment (direct from global, aligned float4s) ----
    v16bf af;
    putA(af, arow + k0, hi);

    // ---- 4 WMMA per k-step ----
#pragma unroll
    for (int nt = 0; nt < 4; ++nt) {
      const int nn = nt * 16 + lm;
      v16bf bfr;
#pragma unroll
      for (int e = 0; e < 16; ++e) bfr[e] = (__bf16)sB[hi * 16 + e][nn];
      acc[nt] = wmma_bf16(af, bfr, acc[nt]);
    }
  }

  // ---- epilogue ----
#pragma unroll
  for (int nt = 0; nt < 4; ++nt) {
    const int n = n0 + nt * 16 + lm;
    const float bv = bias ? bias[n] : 0.f;
#pragma unroll
    for (int r = 0; r < 8; ++r) {
      const int m = m0 + r + 8 * hi;
      if (m < M) {
        float v = acc[nt][r] + bv;
        if (ACT == 1) v = swishf_(v);
        const size_t idx = (size_t)m * N + n;
        if (RESID) out[idx] = resid[idx] + resScale * v;
        else       out[idx] = v;
      }
    }
  }
}

// ---------------------------------------------------------------------------
// LayerNorm over D=512 per row; one wave per row, 4 rows per block.
// ---------------------------------------------------------------------------
__global__ __launch_bounds__(128) void k_layernorm(
    const float* __restrict__ in, const float* __restrict__ g,
    const float* __restrict__ b, float* __restrict__ out, int rows, int swish)
{
  const int wave = threadIdx.x >> 5;
  const int lane = threadIdx.x & 31;
  const int row = blockIdx.x * 4 + wave;
  if (row >= rows) return;
  const float* __restrict__ p = in + (size_t)row * Dc;

  float s = 0.f;
#pragma unroll
  for (int i = 0; i < 16; ++i) s += p[lane + i * 32];
#pragma unroll
  for (int o = 16; o > 0; o >>= 1) s += __shfl_xor(s, o, 32);
  const float mean = s * (1.f / Dc);

  float vs = 0.f;
#pragma unroll
  for (int i = 0; i < 16; ++i) {
    const float d = p[lane + i * 32] - mean;
    vs += d * d;
  }
#pragma unroll
  for (int o = 16; o > 0; o >>= 1) vs += __shfl_xor(vs, o, 32);
  const float inv = rsqrtf(vs * (1.f / Dc) + 1e-5f);

  float* __restrict__ q = out + (size_t)row * Dc;
#pragma unroll
  for (int i = 0; i < 16; ++i) {
    const int d = lane + i * 32;
    float y = (p[d] - mean) * inv * g[d] + b[d];
    if (swish) y = swishf_(y);
    q[d] = y;
  }
}

// ---------------------------------------------------------------------------
// GLU over channels: in [rows, 2D] -> out [rows, D]: a * sigmoid(g)
// ---------------------------------------------------------------------------
__global__ void k_glu(const float* __restrict__ in, float* __restrict__ out)
{
  const int idx = blockIdx.x * blockDim.x + threadIdx.x;
  const int d = idx & (Dc - 1);
  const int row = idx >> 9;
  const float a = in[(size_t)row * (2 * Dc) + d];
  const float g = in[(size_t)row * (2 * Dc) + Dc + d];
  out[idx] = a * sigmoidf_(g);
}

// ---------------------------------------------------------------------------
// Depthwise conv, K=31, zero-padded in T; in/out [B,T,D]; w [D,31]; + bias.
// ---------------------------------------------------------------------------
__global__ void k_dwconv(const float* __restrict__ in, const float* __restrict__ w,
                         const float* __restrict__ bias, float* __restrict__ out)
{
  const int idx = blockIdx.x * blockDim.x + threadIdx.x;
  const int d = idx & (Dc - 1);
  const int t = (idx >> 9) & (Tc - 1);
  const int b = idx >> 18;
  float s = bias[d];
#pragma unroll
  for (int k = 0; k < KWc; ++k) {
    const int tt = t + k - PADW;
    if (tt >= 0 && tt < Tc)
      s += in[(((size_t)b * Tc + tt) << 9) + d] * w[d * KWc + k];
  }
  out[idx] = s;
}

// ---------------------------------------------------------------------------
// Fused rel-pos attention. One wave per (b, h, 16-query-row tile).
//   ac[t,s] = (q[t]+pbu).k[s];  bd[t,s] = (q[t]+pbv).p[s+T-1-t] (shift in index)
//   prob = softmax((ac+bd)/8);  o[t,:] = prob.v
// ---------------------------------------------------------------------------
__global__ __launch_bounds__(32) void k_attn(
    const float* __restrict__ q, const float* __restrict__ k,
    const float* __restrict__ v,
    const float* __restrict__ pbu, const float* __restrict__ pbv, // [H,DK]
    const float* __restrict__ pmat, float* __restrict__ o)
{
  __shared__ float  sAC[16][Tc];       // scores, later probs (32 KB)
  __shared__ __bf16 sBD[16][528];      // shifted-window bd (16.5 KB)

  const int lane = threadIdx.x;
  const int hi = lane >> 4;
  const int lm = lane & 15;
  const int bh = blockIdx.x;
  const int b = bh >> 3;
  const int h = bh & 7;
  const int t0 = blockIdx.y * 16;
  const float scale = 0.125f;          // 1/sqrt(64)

  // --- q-tile A-fragments (loaded once, reused for all s/j tiles) ----------
  v16bf aqu[2], aqv[2];
  {
    const float* __restrict__ qr =
        q + ((size_t)b * Tc + (t0 + lm)) * Dc + h * DKc;
    const float* __restrict__ ub = pbu + h * DKc;
    const float* __restrict__ wb = pbv + h * DKc;
#pragma unroll
    for (int c = 0; c < 2; ++c) {
      const int base0 = c * 32 + hi * 16;
      const int base1 = c * 32 + hi * 16 + 8;
#pragma unroll
      for (int e = 0; e < 8; ++e) {
        const float q0 = qr[base0 + e];
        const float q1 = qr[base1 + e];
        aqu[c][e]     = (__bf16)(q0 + ub[base0 + e]);
        aqu[c][8 + e] = (__bf16)(q1 + ub[base1 + e]);
        aqv[c][e]     = (__bf16)(q0 + wb[base0 + e]);
        aqv[c][8 + e] = (__bf16)(q1 + wb[base1 + e]);
      }
    }
  }

  // --- content scores ac: 32 s-tiles ----------------------------------------
  const float* __restrict__ kb = k + (size_t)b * Tc * Dc + h * DKc;
  for (int st = 0; st < 32; ++st) {
    v8f acc = {};
    const float* __restrict__ krow = kb + (size_t)(st * 16 + lm) * Dc;
#pragma unroll
    for (int c = 0; c < 2; ++c) {
      v16bf bfr;
      put16(bfr, krow + c * 32 + hi * 16);
      acc = wmma_bf16(aqu[c], bfr, acc);
    }
#pragma unroll
    for (int r = 0; r < 8; ++r) sAC[r + 8 * hi][st * 16 + lm] = acc[r];
  }

  // --- position scores bd over shifted window: 33 j-tiles --------------------
  const int jbase = (Tc - 1) - t0 - 15;      // j = jbase + jj
  for (int jt = 0; jt < 33; ++jt) {
    v8f acc = {};
    const int j = jbase + jt * 16 + lm;
    const float* __restrict__ pr = pmat + (size_t)j * Dc + h * DKc;
#pragma unroll
    for (int c = 0; c < 2; ++c) {
      v16bf bfr;
      put16(bfr, pr + c * 32 + hi * 16);
      acc = wmma_bf16(aqv[c], bfr, acc);
    }
#pragma unroll
    for (int r = 0; r < 8; ++r)
      sBD[r + 8 * hi][jt * 16 + lm] = (__bf16)acc[r];
  }

  // --- softmax row-by-row (lane-parallel over s) -----------------------------
  for (int t = 0; t < 16; ++t) {
    float vals[16];
    float mx = -3.0e38f;
#pragma unroll
    for (int i = 0; i < 16; ++i) {
      const int s = lane + i * 32;
      const int jj = s + 15 - t;               // jj in [0, 526]
      const float x = (sAC[t][s] + (float)sBD[t][jj]) * scale;
      vals[i] = x;
      mx = fmaxf(mx, x);
    }
#pragma unroll
    for (int of = 16; of > 0; of >>= 1) mx = fmaxf(mx, __shfl_xor(mx, of, 32));
    float sum = 0.f;
#pragma unroll
    for (int i = 0; i < 16; ++i) {
      const float e = __expf(vals[i] - mx);
      vals[i] = e;
      sum += e;
    }
#pragma unroll
    for (int of = 16; of > 0; of >>= 1) sum += __shfl_xor(sum, of, 32);
    const float inv = 1.f / sum;
#pragma unroll
    for (int i = 0; i < 16; ++i) sAC[t][lane + i * 32] = vals[i] * inv;
  }

  // --- o = prob . v ----------------------------------------------------------
  const float* __restrict__ vrows = v + (size_t)b * Tc * Dc + h * DKc;
  v8f oacc[4] = {v8f{}, v8f{}, v8f{}, v8f{}};
  for (int kc = 0; kc < 16; ++kc) {            // 32-wide s chunks
    v16bf ap;
    {
      const float* p0 = &sAC[lm][kc * 32 + hi * 16];
      put4(ap, 0, ld4(p0)); put4(ap, 4, ld4(p0 + 4));
      const float* p1 = &sAC[lm][kc * 32 + hi * 16 + 8];
      put4(ap, 8, ld4(p1)); put4(ap, 12, ld4(p1 + 4));
    }
#pragma unroll
    for (int nt = 0; nt < 4; ++nt) {
      v16bf bfr;
#pragma unroll
      for (int e = 0; e < 16; ++e) {
        const int ss = kc * 32 + hi * 16 + e;
        bfr[e] = (__bf16)vrows[(size_t)ss * Dc + nt * 16 + lm];
      }
      oacc[nt] = wmma_bf16(ap, bfr, oacc[nt]);
    }
  }
  float* __restrict__ ob = o + ((size_t)b * Tc + t0) * Dc + h * DKc;
#pragma unroll
  for (int nt = 0; nt < 4; ++nt)
#pragma unroll
    for (int r = 0; r < 8; ++r)
      ob[(size_t)(r + 8 * hi) * Dc + nt * 16 + lm] = oacc[nt][r];
}

// ---------------------------------------------------------------------------
// Host orchestration
// ---------------------------------------------------------------------------
extern "C" void kernel_launch(void* const* d_in, const int* in_sizes, int n_in,
                              void* d_out, int out_size, void* d_ws, size_t ws_size,
                              hipStream_t stream)
{
  int ii = 0;
  const float* x_in    = (const float*)d_in[ii++];
  const float* pos_emb = (const float*)d_in[ii++];
  const float* ln1_g   = (const float*)d_in[ii++];
  const float* ln1_b   = (const float*)d_in[ii++];
  const float* ff1_w1  = (const float*)d_in[ii++];
  const float* ff1_b1  = (const float*)d_in[ii++];
  const float* ff1_w2  = (const float*)d_in[ii++];
  const float* ff1_b2  = (const float*)d_in[ii++];
  const float* lnA_g   = (const float*)d_in[ii++];
  const float* lnA_b   = (const float*)d_in[ii++];
  const float* Wq      = (const float*)d_in[ii++];
  const float* bq      = (const float*)d_in[ii++];
  const float* Wk      = (const float*)d_in[ii++];
  const float* bk      = (const float*)d_in[ii++];
  const float* Wv      = (const float*)d_in[ii++];
  const float* bv      = (const float*)d_in[ii++];
  const float* Wo      = (const float*)d_in[ii++];
  const float* bo      = (const float*)d_in[ii++];
  const float* Wpos    = (const float*)d_in[ii++];
  const float* pbu     = (const float*)d_in[ii++];
  const float* pbv     = (const float*)d_in[ii++];
  const float* lnC_g   = (const float*)d_in[ii++];
  const float* lnC_b   = (const float*)d_in[ii++];
  const float* pw1_w   = (const float*)d_in[ii++];
  const float* pw1_b   = (const float*)d_in[ii++];
  const float* dw_w    = (const float*)d_in[ii++];
  const float* dw_b    = (const float*)d_in[ii++];
  const float* cln_g   = (const float*)d_in[ii++];
  const float* cln_b   = (const float*)d_in[ii++];
  const float* pw2_w   = (const float*)d_in[ii++];
  const float* pw2_b   = (const float*)d_in[ii++];
  const float* ln2_g   = (const float*)d_in[ii++];
  const float* ln2_b   = (const float*)d_in[ii++];
  const float* ff2_w1  = (const float*)d_in[ii++];
  const float* ff2_b1  = (const float*)d_in[ii++];
  const float* ff2_w2  = (const float*)d_in[ii++];
  const float* ff2_b2  = (const float*)d_in[ii++];
  const float* lnO_g   = (const float*)d_in[ii++];
  const float* lnO_b   = (const float*)d_in[ii++];
  (void)in_sizes; (void)n_in; (void)out_size; (void)ws_size;

  constexpr size_t SZ_ACT = (size_t)ROWS * Dc;          // 2M floats
  float* ws = (float*)d_ws;
  float* A  = ws;                                       // residual stream
  float* Bf = ws + SZ_ACT;                              // LN output
  float* C  = ws + 2 * SZ_ACT;                          // 8M scratch
  float* P  = ws + 6 * SZ_ACT;                          // pos proj (>=1024 rows)
  float* X  = (float*)d_out;                            // running activation

  const dim3 gLN((ROWS + 3) / 4);

  for (int l = 0; l < 4; ++l) {
    const float* x = (l == 0) ? x_in : X;
    const size_t oD  = (size_t)l * Dc;
    const size_t oDD = (size_t)l * Dc * Dc;
    const size_t oDF = (size_t)l * Dc * DFFc;

    // ---- FFN 1 (half-step residual) ----
    k_layernorm<<<gLN, 128, 0, stream>>>(x, ln1_g + oD, ln1_b + oD, Bf, ROWS, 0);
    k_gemm<false, 1, false><<<dim3(ROWS / 64, DFFc / 64), 128, 0, stream>>>(
        Bf, Dc, ff1_w1 + oDF, DFFc, ff1_b1 + (size_t)l * DFFc,
        nullptr, 0.f, C, ROWS, DFFc, Dc);
    k_gemm<false, 0, true><<<dim3(ROWS / 64, Dc / 64), 128, 0, stream>>>(
        C, DFFc, ff1_w2 + oDF, Dc, ff1_b2 + oD, x, 0.5f, A, ROWS, Dc, DFFc);

    // ---- rel-pos MHSA ----
    k_layernorm<<<gLN, 128, 0, stream>>>(A, lnA_g + oD, lnA_b + oD, Bf, ROWS, 0);
    float* Q  = C;
    float* Kt = C + SZ_ACT;
    float* Vt = C + 2 * SZ_ACT;
    float* O  = C + 3 * SZ_ACT;
    k_gemm<false, 0, false><<<dim3(ROWS / 64, Dc / 64), 128, 0, stream>>>(
        Bf, Dc, Wq + oDD, Dc, bq + oD, nullptr, 0.f, Q, ROWS, Dc, Dc);
    k_gemm<false, 0, false><<<dim3(ROWS / 64, Dc / 64), 128, 0, stream>>>(
        Bf, Dc, Wk + oDD, Dc, bk + oD, nullptr, 0.f, Kt, ROWS, Dc, Dc);
    k_gemm<false, 0, false><<<dim3(ROWS / 64, Dc / 64), 128, 0, stream>>>(
        Bf, Dc, Wv + oDD, Dc, bv + oD, nullptr, 0.f, Vt, ROWS, Dc, Dc);
    k_gemm<false, 0, false><<<dim3(16, Dc / 64), 128, 0, stream>>>(
        pos_emb, Dc, Wpos + oDD, Dc, nullptr, nullptr, 0.f, P,
        2 * Tc - 1, Dc, Dc);
    k_attn<<<dim3(Bc * Hc, Tc / 16), 32, 0, stream>>>(
        Q, Kt, Vt, pbu + (size_t)l * Hc * DKc, pbv + (size_t)l * Hc * DKc, P, O);
    k_gemm<false, 0, true><<<dim3(ROWS / 64, Dc / 64), 128, 0, stream>>>(
        O, Dc, Wo + oDD, Dc, bo + oD, A, 1.f, A, ROWS, Dc, Dc);

    // ---- convolution module ----
    k_layernorm<<<gLN, 128, 0, stream>>>(A, lnC_g + oD, lnC_b + oD, Bf, ROWS, 0);
    float* G2 = C;                 // [ROWS, 2D]
    float* GL = C + 2 * SZ_ACT;    // [ROWS, D]
    float* CV = C + 3 * SZ_ACT;    // [ROWS, D]
    k_gemm<true, 0, false><<<dim3(ROWS / 64, (2 * Dc) / 64), 128, 0, stream>>>(
        Bf, Dc, pw1_w + (size_t)l * 2 * Dc * Dc, Dc,
        pw1_b + (size_t)l * 2 * Dc, nullptr, 0.f, G2, ROWS, 2 * Dc, Dc);
    k_glu<<<(ROWS * Dc) / 256, 256, 0, stream>>>(G2, GL);
    k_dwconv<<<(ROWS * Dc) / 256, 256, 0, stream>>>(
        GL, dw_w + (size_t)l * Dc * KWc, dw_b + oD, CV);
    k_layernorm<<<gLN, 128, 0, stream>>>(CV, cln_g + oD, cln_b + oD, Bf, ROWS, 1);
    k_gemm<true, 0, true><<<dim3(ROWS / 64, Dc / 64), 128, 0, stream>>>(
        Bf, Dc, pw2_w + oDD, Dc, pw2_b + oD, A, 1.f, A, ROWS, Dc, Dc);

    // ---- FFN 2 (half-step residual) ----
    k_layernorm<<<gLN, 128, 0, stream>>>(A, ln2_g + oD, ln2_b + oD, Bf, ROWS, 0);
    k_gemm<false, 1, false><<<dim3(ROWS / 64, DFFc / 64), 128, 0, stream>>>(
        Bf, Dc, ff2_w1 + oDF, DFFc, ff2_b1 + (size_t)l * DFFc,
        nullptr, 0.f, C, ROWS, DFFc, Dc);
    k_gemm<false, 0, true><<<dim3(ROWS / 64, Dc / 64), 128, 0, stream>>>(
        C, DFFc, ff2_w2 + oDF, Dc, ff2_b2 + oD, A, 0.5f, A, ROWS, Dc, DFFc);

    // ---- output LN ----
    k_layernorm<<<gLN, 128, 0, stream>>>(A, lnO_g + oD, lnO_b + oD, X, ROWS, 0);
  }
}
